// CrossAttention_41592463294467
// MI455X (gfx1250) — matrix operand order repaired and checked by reading, not compiled, not run on hardware
//
#include <hip/hip_runtime.h>
#include <stdint.h>

// ---------------- problem constants ----------------
#define Bb 8
#define Nn 1024
#define Cc 768
#define Hh 8
#define Dd 96
#define SCALE 0.10206207261596577f   // 96^-0.5

typedef __attribute__((ext_vector_type(16))) __bf16 bf16x16;
typedef __attribute__((ext_vector_type(8)))  __bf16 bf16x8;
typedef __attribute__((ext_vector_type(8)))  float  f32x8;

#if __has_builtin(__builtin_amdgcn_global_load_async_to_lds_b128)
#define USE_ASYNC 1
#else
#define USE_ASYNC 0
#endif

#if USE_ASYNC
// builtin prototype (from hipcc diagnostic): (v4i addrspace(1)*, v4i addrspace(3)*, imm offset, imm cpol)
typedef int v4i_vs __attribute__((vector_size(16)));
typedef __attribute__((address_space(1))) v4i_vs* gbl_v4i_p;
typedef __attribute__((address_space(3))) v4i_vs* lds_v4i_p;

// async 16-byte global -> LDS copy (per-lane), tracked by ASYNCcnt
__device__ __forceinline__ void async_cp16(void* lds, const void* g) {
    __builtin_amdgcn_global_load_async_to_lds_b128(
        (gbl_v4i_p)(g), (lds_v4i_p)(lds), 0, 0);
}
__device__ __forceinline__ void wait_async0() {
#if __has_builtin(__builtin_amdgcn_s_wait_asynccnt)
    __builtin_amdgcn_s_wait_asynccnt(0);
#else
    asm volatile("s_wait_asynccnt 0x0" ::: "memory");
#endif
}
#endif

// fp32 -> bf16 bits, round-to-nearest-even (pure integer, no __bf16 scalar math)
__device__ __forceinline__ unsigned short f2bf(float f) {
    union { float f; uint32_t u; } a; a.f = f;
    uint32_t r = a.u + 0x7FFFu + ((a.u >> 16) & 1u);
    return (unsigned short)(r >> 16);
}

// ---- WMMA fragment loaders (LDS tiles stored as ushort bf16 bits) ----
// A-matrix 16x32 bf16: lane(l<16)=row M=l holds K {0..7,16..23}; lanes 16..31 hold K {8..15,24..31}
__device__ __forceinline__ bf16x16 frag_a(const unsigned short* base, int stride,
                                          int m, int k0, int hi) {
    union { bf16x16 v; bf16x8 h[2]; } u;
    const unsigned short* p = base + m * stride + k0 + hi * 8;
    u.h[0] = *(const bf16x8*)(p);
    u.h[1] = *(const bf16x8*)(p + 16);
    return u.v;
}
// B-matrix 32x16 bf16: lane holds column n = lane&15; lanes<16: K {0..7,16..23}, lanes>=16: K {8..15,24..31}
__device__ __forceinline__ bf16x16 frag_b(const unsigned short* base, int stride,
                                          int n, int k0, int hi) {
    union { bf16x16 v; bf16x8 h[2]; } u;
    const unsigned short* p = base + n * stride + k0 + hi * 8;
    u.h[0] = *(const bf16x8*)(p);
    u.h[1] = *(const bf16x8*)(p + 16);
    return u.v;
}

__device__ __forceinline__ float redmax16(float v) {
    v = fmaxf(v, __shfl_xor(v, 1));
    v = fmaxf(v, __shfl_xor(v, 2));
    v = fmaxf(v, __shfl_xor(v, 4));
    v = fmaxf(v, __shfl_xor(v, 8));
    return v;
}
__device__ __forceinline__ float redsum16(float v) {
    v += __shfl_xor(v, 1);
    v += __shfl_xor(v, 2);
    v += __shfl_xor(v, 4);
    v += __shfl_xor(v, 8);
    return v;
}

// =====================================================================
// Kernel 1: fused Q/KV projections.  grid = B * (N/64) * (2304/32)
// out tiles 64(n) x 32(o), 8 waves arranged 4x2, K-step 32, 24 steps.
// (fp32 inputs need convert-on-stage, so no async path here)
// =====================================================================
__global__ __launch_bounds__(256) void proj_qkv(
    const float* __restrict__ x, const float* __restrict__ kvin,
    const float* __restrict__ Wq, const float* __restrict__ Wkv,
    unsigned short* __restrict__ Qbf, unsigned short* __restrict__ Kbf,
    unsigned short* __restrict__ Vbf)
{
    const int blk = blockIdx.x;
    const int ot = blk % 72;
    const int mt = (blk / 72) % 16;
    const int b  = blk / (72 * 16);
    const int o0 = ot * 32;
    const bool isq = (o0 < Cc);

    const float* A = isq ? x : kvin;
    const float* W = isq ? (Wq + (size_t)o0 * Cc) : (Wkv + (size_t)(o0 - Cc) * Cc);
    const float* Arow0 = A + ((size_t)b * Nn + mt * 64) * Cc;

    __shared__ __align__(16) unsigned short As[64][32];
    __shared__ __align__(16) unsigned short Bs[32][32];

    const int tid  = threadIdx.x;
    const int wave = tid >> 5, lane = tid & 31;
    const int hi = lane >> 4, ln = lane & 15;
    const int wm = (wave & 3) * 16;      // 0,16,32,48
    const int wn = (wave >> 2) * 16;     // 0,16

    f32x8 acc = {0.f, 0.f, 0.f, 0.f, 0.f, 0.f, 0.f, 0.f};

    for (int kk = 0; kk < Cc; kk += 32) {
        {   // stage A: 64x32, 8 elems/thread
            const int row = tid >> 2;
            const int k0  = (tid & 3) * 8;
            const float* src = Arow0 + (size_t)row * Cc + kk + k0;
            #pragma unroll
            for (int j = 0; j < 8; ++j) As[row][k0 + j] = f2bf(src[j]);
        }
        {   // stage B: 32x32, 4 elems/thread (W rows are K-contiguous)
            const int row = tid >> 3;
            const int k0  = (tid & 7) * 4;
            const float* src = W + (size_t)row * Cc + kk + k0;
            #pragma unroll
            for (int j = 0; j < 4; ++j) Bs[row][k0 + j] = f2bf(src[j]);
        }
        __syncthreads();
        bf16x16 af = frag_a(&As[wm][0], 32, ln, 0, hi);
        bf16x16 bf = frag_b(&Bs[wn][0], 32, ln, 0, hi);
        acc = __builtin_amdgcn_wmma_f32_16x16x32_bf16(false, af, false, bf,
                                                      (short)0, acc, false, false);
        __syncthreads();
    }

    // epilogue: scatter into attention layouts (bf16)
    #pragma unroll
    for (int r = 0; r < 8; ++r) {
        const int n = mt * 64 + wm + r + 8 * hi;   // global row in [0,1024)
        const int o = o0 + wn + ln;                // global out col
        float v = acc[r];
        if (isq) {
            v *= SCALE;
            // reference reshape: (B,N,C)->(B,H,N,D) without transpose
            const int h  = n >> 7;                       // n/128
            const int n2 = ((n & 127) << 3) + (o / Dd);  // (n%128)*8 + c/96
            const int d  = o % Dd;
            Qbf[(((size_t)b * Hh + h) * Nn + n2) * Dd + d] = f2bf(v);
        } else {
            const int oo  = o - Cc;          // [0,1536)
            const int s   = oo / Cc;         // 0 = K, 1 = V
            const int rem = oo % Cc;
            const int h   = rem / Dd;
            const int d   = rem % Dd;
            unsigned short* dst = (s == 0) ? Kbf : Vbf;
            dst[(((size_t)b * Hh + h) * Nn + n) * Dd + d] = f2bf(v);
        }
    }
}

// =====================================================================
// Kernel 2: flash attention.  grid = B*H*8 ; 256 threads = 8 waves.
// WG handles 128 query rows of one (b,h); wave handles 16 rows.
// Key tiles of 32; S: 6 WMMAs, PV: 6 WMMAs per tile; online softmax.
// K/V/Q staged with GLOBAL_LOAD_ASYNC_TO_LDS_B128 (ASYNCcnt) when available.
// =====================================================================
__global__ __launch_bounds__(256) void attn(
    const unsigned short* __restrict__ Qg, const unsigned short* __restrict__ Kg,
    const unsigned short* __restrict__ Vg, unsigned short* __restrict__ Og)
{
    const int qblk = blockIdx.x & 7;
    const int bh   = blockIdx.x >> 3;
    const int b = bh / Hh, h = bh % Hh;

    __shared__ __align__(16) unsigned short Qs[128][Dd];   // 24 KB
    __shared__ __align__(16) unsigned short Ks[32][Dd];    // 6 KB
    __shared__ __align__(16) unsigned short Vs[32][Dd];    // 6 KB (row-major landing pad)
    __shared__ __align__(16) unsigned short Vt[Dd][32];    // 6 KB (transposed)
    __shared__ __align__(16) unsigned short Ps[8][16][32]; // 8 KB

    const int tid  = threadIdx.x;
    const int wave = tid >> 5, lane = tid & 31;
    const int hi = lane >> 4, ln = lane & 15;

    const unsigned short* qsrc = Qg + ((size_t)bh * Nn + qblk * 128) * Dd;
    // stage all 128 Q rows once
#if USE_ASYNC
    {
        char* dst = (char*)&Qs[0][0];
        const char* src = (const char*)qsrc;
        for (int i = tid; i < (128 * Dd * 2) / 16; i += 256)
            async_cp16(dst + i * 16, src + i * 16);
        wait_async0();
    }
#else
    {
        const uint32_t* src = (const uint32_t*)qsrc;
        uint32_t* dst = (uint32_t*)&Qs[0][0];
        for (int i = tid; i < 128 * Dd / 2; i += 256) dst[i] = src[i];
    }
#endif
    __syncthreads();

    // preload 3 Q A-fragments (K-steps over D=96)
    bf16x16 qf[3];
    #pragma unroll
    for (int ks = 0; ks < 3; ++ks)
        qf[ks] = frag_a(&Qs[wave * 16][0], Dd, ln, ks * 32, hi);

    f32x8 Of[6];
    #pragma unroll
    for (int f = 0; f < 6; ++f) Of[f] = (f32x8){0.f,0.f,0.f,0.f,0.f,0.f,0.f,0.f};
    float mrow[8], lrow[8];
    #pragma unroll
    for (int r = 0; r < 8; ++r) { mrow[r] = -3.0e38f; lrow[r] = 0.f; }

    const unsigned short* kbase = Kg + (size_t)bh * Nn * Dd;
    const unsigned short* vbase = Vg + (size_t)bh * Nn * Dd;

    for (int kt = 0; kt < Nn / 32; ++kt) {
        __syncthreads();   // previous tile fully consumed
#if USE_ASYNC
        {   // async stage K tile -> Ks and V tile -> Vs (row-major), 16B chunks
            const char* gk = (const char*)(kbase + (size_t)kt * 32 * Dd);
            const char* gv = (const char*)(vbase + (size_t)kt * 32 * Dd);
            char* dk = (char*)&Ks[0][0];
            char* dv = (char*)&Vs[0][0];
            for (int i = tid; i < (32 * Dd * 2) / 16; i += 256) {
                async_cp16(dk + i * 16, gk + i * 16);
                async_cp16(dv + i * 16, gv + i * 16);
            }
            wait_async0();
        }
        __syncthreads();
        {   // LDS->LDS transpose Vs -> Vt[d][key]
            const uint32_t* vs32 = (const uint32_t*)&Vs[0][0];
            for (int i = tid; i < 32 * Dd / 2; i += 256) {
                const int key = i / (Dd / 2);
                const int d2  = (i % (Dd / 2)) * 2;
                union { uint32_t u; unsigned short s[2]; } w; w.u = vs32[i];
                Vt[d2][key]     = w.s[0];
                Vt[d2 + 1][key] = w.s[1];
            }
        }
#else
        {   // stage K tile (row-major, K-contiguous)
            const uint32_t* src = (const uint32_t*)(kbase + (size_t)kt * 32 * Dd);
            uint32_t* dst = (uint32_t*)&Ks[0][0];
            for (int i = tid; i < 32 * Dd / 2; i += 256) dst[i] = src[i];
        }
        {   // stage V tile transposed: Vt[d][key]
            const uint32_t* src = (const uint32_t*)(vbase + (size_t)kt * 32 * Dd);
            for (int i = tid; i < 32 * Dd / 2; i += 256) {
                const int key = i / (Dd / 2);
                const int d2  = (i % (Dd / 2)) * 2;
                union { uint32_t u; unsigned short s[2]; } w; w.u = src[i];
                Vt[d2][key]     = w.s[0];
                Vt[d2 + 1][key] = w.s[1];
            }
        }
        if (kt + 1 < Nn / 32 && tid < 96) {
            __builtin_prefetch((const char*)(kbase + (size_t)(kt + 1) * 32 * Dd) + tid * 64, 0, 1);
            __builtin_prefetch((const char*)(vbase + (size_t)(kt + 1) * 32 * Dd) + tid * 64, 0, 1);
        }
#endif
        __syncthreads();

        // S = Q * K^T  (16 x 32 in two accumulators)
        f32x8 S0 = (f32x8){0.f,0.f,0.f,0.f,0.f,0.f,0.f,0.f};
        f32x8 S1 = (f32x8){0.f,0.f,0.f,0.f,0.f,0.f,0.f,0.f};
        #pragma unroll
        for (int ks = 0; ks < 3; ++ks) {
            bf16x16 kb0 = frag_b(&Ks[0][0],  Dd, ln, ks * 32, hi);
            bf16x16 kb1 = frag_b(&Ks[16][0], Dd, ln, ks * 32, hi);
            S0 = __builtin_amdgcn_wmma_f32_16x16x32_bf16(false, qf[ks], false, kb0,
                                                         (short)0, S0, false, false);
            S1 = __builtin_amdgcn_wmma_f32_16x16x32_bf16(false, qf[ks], false, kb1,
                                                         (short)0, S1, false, false);
        }

        // online softmax (row r lives at vgpr r, lanes of same 16-half)
        #pragma unroll
        for (int r = 0; r < 8; ++r) {
            float tmax = fmaxf(redmax16(S0[r]), redmax16(S1[r]));
            float mnew = fmaxf(mrow[r], tmax);
            float alpha = __expf(mrow[r] - mnew);
            float p0 = __expf(S0[r] - mnew);
            float p1 = __expf(S1[r] - mnew);
            S0[r] = p0; S1[r] = p1;
            float rs = redsum16(p0) + redsum16(p1);
            lrow[r] = lrow[r] * alpha + rs;
            mrow[r] = mnew;
            #pragma unroll
            for (int f = 0; f < 6; ++f) Of[f][r] *= alpha;
        }

        // P (fp32 C-layout) -> LDS bf16 -> reload as A fragment (same-wave LDS is in-order)
        #pragma unroll
        for (int r = 0; r < 8; ++r) {
            const int m = r + 8 * hi;
            Ps[wave][m][ln]      = f2bf(S0[r]);
            Ps[wave][m][16 + ln] = f2bf(S1[r]);
        }
        bf16x16 pa = frag_a(&Ps[wave][0][0], 32, ln, 0, hi);

        // O += P (16x32 keys) x V (32 x 96)
        #pragma unroll
        for (int f = 0; f < 6; ++f) {
            bf16x16 vb = frag_b(&Vt[f * 16][0], 32, ln, 0, hi);
            Of[f] = __builtin_amdgcn_wmma_f32_16x16x32_bf16(false, pa, false, vb,
                                                            (short)0, Of[f], false, false);
        }
    }

    // finalize and write O in [B,N,C] order (bf16) for the output projection
    float inv[8];
    #pragma unroll
    for (int r = 0; r < 8; ++r) inv[r] = 1.0f / lrow[r];
    #pragma unroll
    for (int f = 0; f < 6; ++f) {
        #pragma unroll
        for (int r = 0; r < 8; ++r) {
            const int n2 = qblk * 128 + wave * 16 + r + 8 * hi;
            const int c  = h * Dd + f * 16 + ln;
            Og[((size_t)b * Nn + n2) * Cc + c] = f2bf(Of[f][r] * inv[r]);
        }
    }
}

// =====================================================================
// Kernel 3: output projection + bias.  grid = B * (N/64) * (C/32)
// A-tile is already bf16 -> async copy straight into LDS.
// =====================================================================
__global__ __launch_bounds__(256) void proj_out(
    const unsigned short* __restrict__ Ob, const float* __restrict__ Wp,
    const float* __restrict__ bp, float* __restrict__ out)
{
    const int blk = blockIdx.x;
    const int ot = blk % 24;
    const int mt = (blk / 24) % 16;
    const int b  = blk / (24 * 16);
    const int o0 = ot * 32;

    __shared__ __align__(16) unsigned short As[64][32];
    __shared__ __align__(16) unsigned short Bs[32][32];

    const int tid  = threadIdx.x;
    const int wave = tid >> 5, lane = tid & 31;
    const int hi = lane >> 4, ln = lane & 15;
    const int wm = (wave & 3) * 16;
    const int wn = (wave >> 2) * 16;

    f32x8 acc = {0.f, 0.f, 0.f, 0.f, 0.f, 0.f, 0.f, 0.f};

    for (int kk = 0; kk < Cc; kk += 32) {
        {   // stage A (already bf16): one 16B chunk per thread
            const int row = tid >> 2;
            const int k0  = (tid & 3) * 8;
            const unsigned short* src =
                Ob + ((size_t)b * Nn + mt * 64 + row) * Cc + kk + k0;
#if USE_ASYNC
            async_cp16(&As[row][k0], src);
#else
            *(bf16x8*)&As[row][k0] = *(const bf16x8*)src;
#endif
        }
        {   // stage B from fp32 Wproj
            const int row = tid >> 3;
            const int k0  = (tid & 7) * 4;
            const float* src = Wp + ((size_t)(o0 + row)) * Cc + kk + k0;
            #pragma unroll
            for (int j = 0; j < 4; ++j) Bs[row][k0 + j] = f2bf(src[j]);
        }
#if USE_ASYNC
        wait_async0();
#endif
        __syncthreads();
        bf16x16 af = frag_a(&As[wm][0], 32, ln, 0, hi);
        bf16x16 bf = frag_b(&Bs[wn][0], 32, ln, 0, hi);
        acc = __builtin_amdgcn_wmma_f32_16x16x32_bf16(false, af, false, bf,
                                                      (short)0, acc, false, false);
        __syncthreads();
    }

    #pragma unroll
    for (int r = 0; r < 8; ++r) {
        const int n = mt * 64 + wm + r + 8 * hi;
        const int o = o0 + wn + ln;
        out[((size_t)b * Nn + n) * Cc + o] = acc[r] + bp[o];
    }
}

// =====================================================================
extern "C" void kernel_launch(void* const* d_in, const int* in_sizes, int n_in,
                              void* d_out, int out_size, void* d_ws, size_t ws_size,
                              hipStream_t stream) {
    const float* x     = (const float*)d_in[0];
    const float* kv    = (const float*)d_in[1];
    const float* Wq    = (const float*)d_in[2];
    const float* Wkv   = (const float*)d_in[3];
    const float* Wproj = (const float*)d_in[4];
    const float* bproj = (const float*)d_in[5];
    float* out = (float*)d_out;

    const size_t bhnd = (size_t)Bb * Hh * Nn * Dd * sizeof(unsigned short); // 12.58 MB
    char* ws = (char*)d_ws;
    unsigned short* Qbf = (unsigned short*)(ws);
    unsigned short* Kbf = (unsigned short*)(ws + bhnd);
    unsigned short* Vbf = (unsigned short*)(ws + 2 * bhnd);
    unsigned short* Obf = (unsigned short*)(ws + 3 * bhnd);
    (void)in_sizes; (void)n_in; (void)out_size; (void)ws_size;

    proj_qkv<<<Bb * (Nn / 64) * (2304 / 32), 256, 0, stream>>>(
        x, kv, Wq, Wkv, Qbf, Kbf, Vbf);
    attn<<<Bb * Hh * 8, 256, 0, stream>>>(Qbf, Kbf, Vbf, Obf);
    proj_out<<<Bb * (Nn / 64) * (Cc / 32), 256, 0, stream>>>(
        Obf, Wproj, bproj, out);
}
